// AttentionBlock_10831907521108
// MI455X (gfx1250) — compile-verified, matrix operand order checked by reading, not run on hardware
//
#include <hip/hip_runtime.h>
#include <hip/hip_bf16.h>
#include <cstdint>

typedef __bf16 bf16_t;
typedef __attribute__((ext_vector_type(16))) __bf16 v16bf;
typedef __attribute__((ext_vector_type(8)))  float  v8f;

union Frag16 {
    v16bf v;
    uint4 q[2];
};

#define WMMA_BF16(A, B, C) \
    __builtin_amdgcn_wmma_f32_16x16x32_bf16(false, (A), false, (B), (short)0, (C), false, false)

// ---------------------------------------------------------------------------
// CDNA5 async global->LDS copy helpers (ASYNCcnt-tracked, LDS written directly)
// ISA 10.7: LDS address from VDST, global address from VADDR, both plus the offset
// ---------------------------------------------------------------------------
#define ASYNC_LD16(lds, g, OFF)                                              \
    asm volatile("global_load_async_to_lds_b128 %0, %1, off offset:" #OFF    \
                 :: "v"(lds), "v"(g) : "memory")

__device__ __forceinline__ uint32_t lds_addr(const void* p) {
    return (uint32_t)(uintptr_t)p;
}

__device__ __forceinline__ void async_copy64(uint32_t lds, const void* g) {
    ASYNC_LD16(lds, g, 0);  ASYNC_LD16(lds, g, 16);
    ASYNC_LD16(lds, g, 32); ASYNC_LD16(lds, g, 48);
}

__device__ __forceinline__ void async_copy128(uint32_t lds, const void* g) {
    async_copy64(lds, g);
    ASYNC_LD16(lds, g, 64); ASYNC_LD16(lds, g, 80);
    ASYNC_LD16(lds, g, 96); ASYNC_LD16(lds, g, 112);
}

#define WAIT_ASYNC() asm volatile("s_wait_asynccnt 0" ::: "memory")
#define WAIT_DS()    asm volatile("s_wait_dscnt 0" ::: "memory")

// ---------------------------------------------------------------------------
// Prep kernels: fp32 -> bf16 casts (+ x transpose to token-major)
// ---------------------------------------------------------------------------
__global__ void cast_f32_to_bf16(const float* __restrict__ in, bf16_t* __restrict__ out, int n) {
    int i = blockIdx.x * 256 + threadIdx.x;
    if (i < n) out[i] = (bf16_t)in[i];
}

// x [8,256,1024] -> xtb [b*1024+s][256] (bf16)
__global__ void transpose_cast_x(const float* __restrict__ x, bf16_t* __restrict__ xtb) {
    int i = blockIdx.x * 256 + threadIdx.x;  // 0 .. 8*256*1024
    if (i < 8 * 256 * 1024) {
        int s = i & 1023;
        int c = (i >> 10) & 255;
        int b = i >> 18;
        xtb[(size_t)((b << 10) | s) * 256 + c] = (bf16_t)x[i];
    }
}

// ---------------------------------------------------------------------------
// QKV GEMM: [8192,256] @ [256,6144] + bias -> qkv bf16 [8192,6144]
// Block = 128 threads (4 waves). Block tile 128x64, wave tile 32x64.
// Double-buffered LDS; A tile staged with async global->LDS copies.
// ---------------------------------------------------------------------------
__global__ __launch_bounds__(128) void qkv_gemm(const bf16_t* __restrict__ A,   // xtb [8192,256]
                                                const bf16_t* __restrict__ Bw,  // Wpb [256,6144]
                                                const float*  __restrict__ bias,// bp [6144]
                                                bf16_t* __restrict__ Cq)        // qkv [8192,6144]
{
    const int N = 6144, K = 256;
    __shared__ __attribute__((aligned(16))) bf16_t at[2][128 * 40];  // [m][k] padded
    __shared__ __attribute__((aligned(16))) bf16_t bt[2][64 * 40];   // [n][k] transposed, padded

    const int tid = threadIdx.x;
    const int lane = tid & 31, wave = tid >> 5;
    const int rc = lane & 15;
    const int kb = (lane < 16) ? 0 : 8;
    const int hf = lane >> 4;
    const int n0 = blockIdx.x * 64;
    const int m0 = blockIdx.y * 128;

    const v8f zv = {0.f, 0.f, 0.f, 0.f, 0.f, 0.f, 0.f, 0.f};
    v8f acc[2][4];
#pragma unroll
    for (int mi = 0; mi < 2; mi++)
#pragma unroll
        for (int nj = 0; nj < 4; nj++) acc[mi][nj] = zv;

    auto stage_a = [&](int buf, int k0) {  // 128x32, async, contiguous 64B per lane
        const bf16_t* src = A + (size_t)(m0 + tid) * K + k0;
        async_copy64(lds_addr(&at[buf][tid * 40]), src);
    };
    auto stage_b = [&](int buf, int k0) {  // 32x64 transposed -> bt[n][k]
        int k  = tid >> 2;   // 0..31
        int ng = tid & 3;    // 0..3 -> 16 cols each
        const bf16_t* src = Bw + (size_t)(k0 + k) * N + n0 + ng * 16;
#pragma unroll
        for (int i = 0; i < 16; i++) bt[buf][(ng * 16 + i) * 40 + k] = src[i];
    };

    stage_a(0, 0);
    stage_b(0, 0);
    WAIT_ASYNC();
    __syncthreads();

    int cur = 0;
    for (int k0 = 0; k0 < K; k0 += 32) {
        int nxt = cur ^ 1;
        if (k0 + 32 < K) {  // prefetch next tile while computing on current
            stage_a(nxt, k0 + 32);
            stage_b(nxt, k0 + 32);
        }

#pragma unroll
        for (int mi = 0; mi < 2; mi++) {
            Frag16 fa;
            const bf16_t* ap = &at[cur][(wave * 32 + mi * 16 + rc) * 40 + kb];
            fa.q[0] = *(const uint4*)(ap);
            fa.q[1] = *(const uint4*)(ap + 16);
#pragma unroll
            for (int nj = 0; nj < 4; nj++) {
                Frag16 fb;
                const bf16_t* bp = &bt[cur][(nj * 16 + rc) * 40 + kb];
                fb.q[0] = *(const uint4*)(bp);
                fb.q[1] = *(const uint4*)(bp + 16);
                acc[mi][nj] = WMMA_BF16(fa.v, fb.v, acc[mi][nj]);
            }
        }
        WAIT_ASYNC();
        __syncthreads();
        cur = nxt;
    }

#pragma unroll
    for (int mi = 0; mi < 2; mi++)
#pragma unroll
        for (int nj = 0; nj < 4; nj++) {
            int col = n0 + nj * 16 + rc;
            float bv = bias[col];
#pragma unroll
            for (int p = 0; p < 8; p++) {
                int row = m0 + wave * 32 + mi * 16 + p + 8 * hf;
                Cq[(size_t)row * N + col] = (bf16_t)(acc[mi][nj][p] + bv);
            }
        }
}

// ---------------------------------------------------------------------------
// Flash attention: per (b,h), 64-query blocks; wave owns 16 query rows.
// qkv layout per token row: [h*768 + {q:0..255, k:256..511, v:512..767}]
// K tile staged via async global->LDS; V transposed manually for P@V frags.
// ---------------------------------------------------------------------------
__global__ __launch_bounds__(128) void attn_kernel(const bf16_t* __restrict__ qkv,  // [8192,6144]
                                                   bf16_t* __restrict__ Ob)         // [8192,2048]
{
    const int S = 1024, LD = 6144;
    __shared__ __attribute__((aligned(16))) bf16_t kt[32 * 264];   // [key][dk] padded
    __shared__ __attribute__((aligned(16))) bf16_t vt[256 * 40];   // [dk][key] transposed padded
    __shared__ __attribute__((aligned(16))) bf16_t pt[4][16 * 40]; // per-wave P staging

    const int tid = threadIdx.x, lane = tid & 31, wave = tid >> 5;
    const int rc = lane & 15, hf = lane >> 4;
    const int kb = hf ? 8 : 0;
    const int bh = blockIdx.y;              // b*8 + h
    const int b = bh >> 3, h = bh & 7;
    const int q0 = blockIdx.x * 64;
    const size_t rowbase = (size_t)b * S;

    const bf16_t* qptr  = qkv + (rowbase + q0) * (size_t)LD + h * 768;
    const bf16_t* kptr0 = qkv + rowbase * (size_t)LD + h * 768 + 256;
    const bf16_t* vptr0 = kptr0 + 256;

    // Q fragments for this wave's 16 rows (A-layout, dk kb..kb+7 then kb+16..kb+23)
    v16bf qf[8];
    {
        const bf16_t* qrow = qptr + (size_t)(wave * 16 + rc) * LD;
#pragma unroll
        for (int kk = 0; kk < 8; kk++) {
            Frag16 f;
            f.q[0] = *(const uint4*)(qrow + kk * 32 + kb);
            f.q[1] = *(const uint4*)(qrow + kk * 32 + kb + 16);
            qf[kk] = f.v;
        }
    }

    const v8f zv = {0.f, 0.f, 0.f, 0.f, 0.f, 0.f, 0.f, 0.f};
    v8f o[16];
#pragma unroll
    for (int d = 0; d < 16; d++) o[d] = zv;
    float m[8], l[8];
#pragma unroll
    for (int p = 0; p < 8; p++) { m[p] = -3.0e38f; l[p] = 0.f; }

    const float SOFT_SCALE = 0.0625f;         // 256^-0.5
    const float L2E = 1.4426950408889634f;

    for (int key0 = 0; key0 < S; key0 += 32) {
        // cooperative stage: K [key][dk] via async copies, V transposed [dk][key]
        {
            int krow = tid >> 2;    // 0..31
            int seg  = tid & 3;     // 64 dk each
            const bf16_t* ks = kptr0 + (size_t)(key0 + krow) * LD + seg * 64;
            const bf16_t* vs = vptr0 + (size_t)(key0 + krow) * LD + seg * 64;
            async_copy128(lds_addr(&kt[krow * 264 + seg * 64]), ks);
#pragma unroll
            for (int i = 0; i < 8; i++) {
                uint4 d4 = *(const uint4*)(vs + i * 8);
                const bf16_t* e = (const bf16_t*)&d4;
#pragma unroll
                for (int j = 0; j < 8; j++) vt[(seg * 64 + i * 8 + j) * 40 + krow] = e[j];
            }
        }
        WAIT_ASYNC();
        __syncthreads();

        // scores S = Q @ K^T : 16 rows x 32 keys (two 16x16 accumulators)
        v8f sc[2];
        sc[0] = zv; sc[1] = zv;
#pragma unroll
        for (int kk = 0; kk < 8; kk++) {
#pragma unroll
            for (int t = 0; t < 2; t++) {
                Frag16 fb;
                const bf16_t* bp = &kt[(t * 16 + rc) * 264 + kk * 32 + kb];
                fb.q[0] = *(const uint4*)(bp);
                fb.q[1] = *(const uint4*)(bp + 16);
                sc[t] = WMMA_BF16(qf[kk], fb.v, sc[t]);
            }
        }

        // online softmax (rows = p + 8*hf, cols across lanes in 16-lane halves)
        float alpha[8];
#pragma unroll
        for (int p = 0; p < 8; p++) {
            float s0 = sc[0][p] * SOFT_SCALE;
            float s1 = sc[1][p] * SOFT_SCALE;
            float mx = fmaxf(s0, s1);
#pragma unroll
            for (int off = 8; off >= 1; off >>= 1) mx = fmaxf(mx, __shfl_xor(mx, off, 32));
            float mn = fmaxf(m[p], mx);
            alpha[p] = exp2f((m[p] - mn) * L2E);
            m[p] = mn;
            float p0 = exp2f((s0 - mn) * L2E);
            float p1 = exp2f((s1 - mn) * L2E);
            sc[0][p] = p0; sc[1][p] = p1;
            float rs = p0 + p1;
#pragma unroll
            for (int off = 8; off >= 1; off >>= 1) rs += __shfl_xor(rs, off, 32);
            l[p] = l[p] * alpha[p] + rs;
        }

        // rescale O accumulators
#pragma unroll
        for (int d = 0; d < 16; d++)
#pragma unroll
            for (int p = 0; p < 8; p++) o[d][p] *= alpha[p];

        // spill P (C-layout) -> per-wave LDS -> reload in A-fragment layout
#pragma unroll
        for (int t = 0; t < 2; t++)
#pragma unroll
            for (int p = 0; p < 8; p++)
                pt[wave][(p + 8 * hf) * 40 + t * 16 + rc] = (bf16_t)sc[t][p];
        WAIT_DS();

        Frag16 fp;
        {
            const bf16_t* pp = &pt[wave][rc * 40 + kb];
            fp.q[0] = *(const uint4*)(pp);
            fp.q[1] = *(const uint4*)(pp + 16);
        }

        // O += P @ V  (16x32 @ 32x256, per 16-wide dk tile)
#pragma unroll
        for (int d = 0; d < 16; d++) {
            Frag16 fv;
            const bf16_t* vp = &vt[(d * 16 + rc) * 40 + kb];
            fv.q[0] = *(const uint4*)(vp);
            fv.q[1] = *(const uint4*)(vp + 16);
            o[d] = WMMA_BF16(fp.v, fv.v, o[d]);
        }
        __syncthreads();
    }

    // normalize + store bf16 O [8192,2048]
#pragma unroll
    for (int p = 0; p < 8; p++) l[p] = 1.0f / l[p];
    const size_t orow0 = rowbase + q0 + wave * 16;
#pragma unroll
    for (int d = 0; d < 16; d++) {
        int col = h * 256 + d * 16 + rc;
#pragma unroll
        for (int p = 0; p < 8; p++) {
            size_t row = orow0 + p + 8 * hf;
            Ob[row * 2048 + col] = (bf16_t)(o[d][p] * l[p]);
        }
    }
}

// ---------------------------------------------------------------------------
// Output GEMM + bias + residual + transpose store:
// [8192,2048] @ [2048,256] + bo + x  -> out [8,256,1024] fp32
// Double-buffered LDS; A tile staged with async global->LDS copies.
// ---------------------------------------------------------------------------
__global__ __launch_bounds__(128) void out_gemm(const bf16_t* __restrict__ A,   // Ob [8192,2048]
                                                const bf16_t* __restrict__ Bw,  // Wob [2048,256]
                                                const float*  __restrict__ bias,// bo [256]
                                                const float*  __restrict__ x,   // [8,256,1024]
                                                float* __restrict__ out)        // [8,256,1024]
{
    const int N = 256, K = 2048;
    __shared__ __attribute__((aligned(16))) bf16_t at[2][128 * 40];
    __shared__ __attribute__((aligned(16))) bf16_t bt[2][64 * 40];

    const int tid = threadIdx.x;
    const int lane = tid & 31, wave = tid >> 5;
    const int rc = lane & 15;
    const int kb = (lane < 16) ? 0 : 8;
    const int hf = lane >> 4;
    const int n0 = blockIdx.x * 64;
    const int m0 = blockIdx.y * 128;

    const v8f zv = {0.f, 0.f, 0.f, 0.f, 0.f, 0.f, 0.f, 0.f};
    v8f acc[2][4];
#pragma unroll
    for (int mi = 0; mi < 2; mi++)
#pragma unroll
        for (int nj = 0; nj < 4; nj++) acc[mi][nj] = zv;

    auto stage_a = [&](int buf, int k0) {
        const bf16_t* src = A + (size_t)(m0 + tid) * K + k0;
        async_copy64(lds_addr(&at[buf][tid * 40]), src);
    };
    auto stage_b = [&](int buf, int k0) {
        int k  = tid >> 2;
        int ng = tid & 3;
        const bf16_t* src = Bw + (size_t)(k0 + k) * N + n0 + ng * 16;
#pragma unroll
        for (int i = 0; i < 16; i++) bt[buf][(ng * 16 + i) * 40 + k] = src[i];
    };

    stage_a(0, 0);
    stage_b(0, 0);
    WAIT_ASYNC();
    __syncthreads();

    int cur = 0;
    for (int k0 = 0; k0 < K; k0 += 32) {
        int nxt = cur ^ 1;
        if (k0 + 32 < K) {
            stage_a(nxt, k0 + 32);
            stage_b(nxt, k0 + 32);
        }

#pragma unroll
        for (int mi = 0; mi < 2; mi++) {
            Frag16 fa;
            const bf16_t* ap = &at[cur][(wave * 32 + mi * 16 + rc) * 40 + kb];
            fa.q[0] = *(const uint4*)(ap);
            fa.q[1] = *(const uint4*)(ap + 16);
#pragma unroll
            for (int nj = 0; nj < 4; nj++) {
                Frag16 fb;
                const bf16_t* bp = &bt[cur][(nj * 16 + rc) * 40 + kb];
                fb.q[0] = *(const uint4*)(bp);
                fb.q[1] = *(const uint4*)(bp + 16);
                acc[mi][nj] = WMMA_BF16(fa.v, fb.v, acc[mi][nj]);
            }
        }
        WAIT_ASYNC();
        __syncthreads();
        cur = nxt;
    }

#pragma unroll
    for (int mi = 0; mi < 2; mi++)
#pragma unroll
        for (int nj = 0; nj < 4; nj++) {
            int col = n0 + nj * 16 + rc;
            float bv = bias[col];
#pragma unroll
            for (int p = 0; p < 8; p++) {
                int row = m0 + wave * 32 + mi * 16 + p + 8 * hf;
                int bb = row >> 10;
                int ss = row & 1023;
                size_t xi = ((size_t)(bb * 256 + col) << 10) + ss;
                out[xi] = acc[mi][nj][p] + bv + x[xi];
            }
        }
}

// ---------------------------------------------------------------------------
// Host launcher
// ---------------------------------------------------------------------------
extern "C" void kernel_launch(void* const* d_in, const int* in_sizes, int n_in,
                              void* d_out, int out_size, void* d_ws, size_t ws_size,
                              hipStream_t stream) {
    (void)in_sizes; (void)n_in; (void)out_size; (void)ws_size;
    const float* x  = (const float*)d_in[0];   // [8,256,32,32]
    const float* Wp = (const float*)d_in[1];   // [256,6144]
    const float* bp = (const float*)d_in[2];   // [6144]
    const float* Wo = (const float*)d_in[3];   // [2048,256]
    const float* bo = (const float*)d_in[4];   // [256]
    float* out = (float*)d_out;

    char* ws = (char*)d_ws;
    bf16_t* xtb = (bf16_t*)(ws + 0);              //  4,194,304 B  [8192,256]
    bf16_t* Wpb = (bf16_t*)(ws + 4194304);        //  3,145,728 B  [256,6144]
    bf16_t* Wob = (bf16_t*)(ws + 7340032);        //  1,048,576 B  [2048,256]
    bf16_t* qkv = (bf16_t*)(ws + 8388608);        // 100,663,296 B [8192,6144]
    bf16_t* Ob  = (bf16_t*)(ws + 109051904);      // 33,554,432 B  [8192,2048]
    // total: 142,606,336 B

    cast_f32_to_bf16<<<6144, 256, 0, stream>>>(Wp, Wpb, 256 * 6144);
    cast_f32_to_bf16<<<2048, 256, 0, stream>>>(Wo, Wob, 2048 * 256);
    transpose_cast_x<<<8192, 256, 0, stream>>>(x, xtb);

    qkv_gemm<<<dim3(96, 64), 128, 0, stream>>>(xtb, Wpb, bp, qkv);
    attn_kernel<<<dim3(16, 64), 128, 0, stream>>>(qkv, Ob);
    out_gemm<<<dim3(4, 64), 128, 0, stream>>>(Ob, Wob, bo, x, out);
}
